// VAE_77472620085272
// MI455X (gfx1250) — matrix operand order repaired and checked by reading, not compiled, not run on hardware
//
#include <hip/hip_runtime.h>
#include <hip/hip_bf16.h>
#include <math.h>

// ---------------------------------------------------------------------------
// CDNA5 WMMA types (wave32): D(16x16 f32) = A(16x32 bf16) x B(32x16 bf16) + C
// ---------------------------------------------------------------------------
typedef __attribute__((ext_vector_type(16))) __bf16        v16bf;
typedef __attribute__((ext_vector_type(8)))  float         v8f;
typedef __attribute__((ext_vector_type(4)))  unsigned int  u32x4;   // POD 16B

union Frag16 { v16bf v; unsigned short u[16]; u32x4 q[2]; };

__device__ __forceinline__ unsigned short f32_to_bf16_rn(float f) {
  unsigned int x = __builtin_bit_cast(unsigned int, f);
  unsigned int r = 0x7FFFu + ((x >> 16) & 1u);
  return (unsigned short)((x + r) >> 16);
}

// ---------------------------------------------------------------------------
// Utility / pack kernels
// ---------------------------------------------------------------------------
__global__ void k_zero_u64(unsigned long long* __restrict__ p, long n) {
  long i = (long)blockIdx.x * blockDim.x + threadIdx.x;
  if (i < n) p[i] = 0ull;
}

__global__ void k_pack_bf16(const float* __restrict__ src,
                            unsigned short* __restrict__ dst, long n) {
  long i = (long)blockIdx.x * blockDim.x + threadIdx.x;
  if (i < n) dst[i] = f32_to_bf16_rn(src[i]);
}

// Conv / ConvT weights -> zero-padded fragment-ready A [Mpad x Kpad] bf16.
// K encoding: k = ci*16 + ky*4 + kx  (ky,kx in 0..2 used; rest zero)
__global__ void k_pack_w_pad(const float* __restrict__ w,
                             unsigned short* __restrict__ A,
                             int co, int ci, int Mpad, int Kpad, int convT) {
  int idx = blockIdx.x * blockDim.x + threadIdx.x;
  int tot = Mpad * Kpad;
  if (idx >= tot) return;
  int m = idx / Kpad;
  int k = idx - m * Kpad;
  int ci_i = k >> 4;
  int ky = (k >> 2) & 3;
  int kx = k & 3;
  float v = 0.f;
  if (m < co && ci_i < ci && ky < 3 && kx < 3) {
    int r = ky * 3 + kx;
    v = convT ? w[((long)ci_i * co + m) * 9 + r]
              : w[((long)m * ci + ci_i) * 9 + r];
  }
  A[idx] = f32_to_bf16_rn(v);
}

// Transpose-pack f32 W[rows x cols] -> bf16 Wt[cols x rows] via LDS tiles.
__global__ void k_transpose_pack(const float* __restrict__ src,
                                 unsigned short* __restrict__ dst,
                                 int rows, int cols) {
  __shared__ unsigned short tile[32][33];
  int k0 = blockIdx.x * 32;
  int n0 = blockIdx.y * 32;
  for (int r = threadIdx.y; r < 32; r += 8) {
    tile[r][threadIdx.x] = f32_to_bf16_rn(src[(long)(k0 + r) * cols + n0 + threadIdx.x]);
  }
  __syncthreads();
  for (int r = threadIdx.y; r < 32; r += 8) {
    dst[(long)(n0 + r) * rows + k0 + threadIdx.x] = tile[threadIdx.x][r];
  }
}

// ---------------------------------------------------------------------------
// WMMA implicit-GEMM. One wave = two adjacent 16x16 C tiles (shared A frag).
//  MODE 0: dense Wt column-major [N][Kpad] -> contiguous b128 B loads
//  MODE 1: conv  k3 s2 p1     : gather offsets precomputed, uniform k-advance
//  MODE 2: convT k3 s2 p1 op1 : same
// Invalid gather elements point into a large zero arena; offsets drift by the
// same uniform step (2*ih*iw) each k-step and stay inside the arena.
// 8 waves/block -> 256 columns. N%256==0, Kpad%32==0, A zero-padded.
// ---------------------------------------------------------------------------
template <int MODE>
__global__ void k_wmma_gemm(const unsigned short* __restrict__ A,
                            const unsigned short* __restrict__ Bsrc,
                            const float* __restrict__ bias, int bias_per_n,
                            float* __restrict__ C,
                            int M, int N, int Kpad,
                            int ci, int ih, int iw, int oh, int ow, int zoff) {
  const int lane  = threadIdx.x & 31;
  const int wave  = threadIdx.x >> 5;
  const int mtile = blockIdx.y;
  const int l15   = lane & 15;
  const int half  = lane >> 4;
  const int hw    = oh * ow;
  const int ihw   = ih * iw;

  const int bn0 = (blockIdx.x * 8 + wave) * 32 + l15;  // tile0 column
  const int bn1 = bn0 + 16;                            // tile1 column

  const long arow = (long)(mtile * 16 + l15) * Kpad;   // A row (padded)
  const int  akb  = half * 16;                         // A K-base per half
  const int  bkh  = half * 16;                         // B K-base per half

  int sp0 = 0, img0 = 0, sp1 = 0, img1 = 0;
  int off0[16], off1[16];
  if constexpr (MODE != 0) {
    img0 = bn0 / hw; sp0 = bn0 - img0 * hw;
    img1 = bn1 / hw; sp1 = bn1 - img1 * hw;
    int oy0 = sp0 / ow, ox0 = sp0 - (sp0 / ow) * ow;
    int oy1 = sp1 / ow, ox1 = sp1 - (sp1 / ow) * ow;
    int base0 = img0 * ci * ihw;
    int base1 = img1 * ci * ihw;
    int oyb0, oxb0, oyb1, oxb1;
    if constexpr (MODE == 1) { oyb0 = oy0 * 2 - 1; oxb0 = ox0 * 2 - 1;
                               oyb1 = oy1 * 2 - 1; oxb1 = ox1 * 2 - 1; }
    else                     { oyb0 = oy0 + 1;     oxb0 = ox0 + 1;
                               oyb1 = oy1 + 1;     oxb1 = ox1 + 1; }
#pragma unroll
    for (int e = 0; e < 16; ++e) {
      int kk = bkh + e;                 // k0 == 0 positions
      int ci0 = kk >> 4;                // 0 or 1
      int ky = (kk >> 2) & 3;
      int kx = kk & 3;
      if constexpr (MODE == 1) {
        int iy0 = oyb0 + ky, ix0 = oxb0 + kx;
        bool v0 = (ky != 3) & (kx != 3) &
                  ((unsigned)iy0 < (unsigned)ih) & ((unsigned)ix0 < (unsigned)iw);
        off0[e] = v0 ? (base0 + ci0 * ihw + iy0 * iw + ix0) : zoff;
        int iy1 = oyb1 + ky, ix1 = oxb1 + kx;
        bool v1 = (ky != 3) & (kx != 3) &
                  ((unsigned)iy1 < (unsigned)ih) & ((unsigned)ix1 < (unsigned)iw);
        off1[e] = v1 ? (base1 + ci0 * ihw + iy1 * iw + ix1) : zoff;
      } else {
        int ty0 = oyb0 - ky, tx0 = oxb0 - kx;
        int iy0 = ty0 >> 1,  ix0 = tx0 >> 1;
        bool v0 = (ky != 3) & (kx != 3) & (((ty0 | tx0) & 1) == 0) &
                  ((unsigned)iy0 < (unsigned)ih) & ((unsigned)ix0 < (unsigned)iw);
        off0[e] = v0 ? (base0 + ci0 * ihw + iy0 * iw + ix0) : zoff;
        int ty1 = oyb1 - ky, tx1 = oxb1 - kx;
        int iy1 = ty1 >> 1,  ix1 = tx1 >> 1;
        bool v1 = (ky != 3) & (kx != 3) & (((ty1 | tx1) & 1) == 0) &
                  ((unsigned)iy1 < (unsigned)ih) & ((unsigned)ix1 < (unsigned)iw);
        off1[e] = v1 ? (base1 + ci0 * ihw + iy1 * iw + ix1) : zoff;
      }
    }
  }

  v8f acc0 = {0.f, 0.f, 0.f, 0.f, 0.f, 0.f, 0.f, 0.f};
  v8f acc1 = {0.f, 0.f, 0.f, 0.f, 0.f, 0.f, 0.f, 0.f};

  const int kstep = 2 * ihw;            // ci advances by 2 per 32-wide k-step

  for (int k0 = 0; k0 < Kpad; k0 += 32) {
    Frag16 a, b0, b1;
    const unsigned short* ap = A + arow + k0 + akb;
    a.q[0] = ((const u32x4*)ap)[0];        // K = akb .. akb+7
    a.q[1] = ((const u32x4*)(ap + 8))[0];  // K = akb+8 .. akb+15
    if constexpr (MODE == 0) {
      const unsigned short* p0 = Bsrc + (long)bn0 * Kpad + k0 + bkh;
      const unsigned short* p1 = Bsrc + (long)bn1 * Kpad + k0 + bkh;
      b0.q[0] = ((const u32x4*)p0)[0];
      b0.q[1] = ((const u32x4*)p0)[1];
      b1.q[0] = ((const u32x4*)p1)[0];
      b1.q[1] = ((const u32x4*)p1)[1];
    } else {
#pragma unroll
      for (int e = 0; e < 16; ++e) b0.u[e] = Bsrc[off0[e]];
#pragma unroll
      for (int e = 0; e < 16; ++e) b1.u[e] = Bsrc[off1[e]];
    }
    acc0 = __builtin_amdgcn_wmma_f32_16x16x32_bf16(false, a.v, false, b0.v, (short)0, acc0, false, false);
    acc1 = __builtin_amdgcn_wmma_f32_16x16x32_bf16(false, a.v, false, b1.v, (short)0, acc1, false, false);
    if constexpr (MODE != 0) {
#pragma unroll
      for (int e = 0; e < 16; ++e) { off0[e] += kstep; off1[e] += kstep; }
    }
  }

  // C/D layout: VGPR r -> row M = r + 8*half, column = lane&15
#pragma unroll
  for (int r = 0; r < 8; ++r) {
    int m = mtile * 16 + half * 8 + r;
    if (m < M) {
      float bv = bias_per_n ? 0.f : bias[m];
      float v0 = acc0[r] + (bias_per_n ? bias[bn0] : bv);
      float v1 = acc1[r] + (bias_per_n ? bias[bn1] : bv);
      if constexpr (MODE == 0) {
        C[(long)m * N + bn0] = v0;
        C[(long)m * N + bn1] = v1;
      } else {
        C[((long)img0 * M + m) * hw + sp0] = v0;
        C[((long)img1 * M + m) * hw + sp1] = v1;
      }
    }
  }
}

// ---------------------------------------------------------------------------
// BatchNorm (training batch stats) — one block per channel, LDS reduce.
// ---------------------------------------------------------------------------
__global__ void k_bn_stats(const float* __restrict__ act,
                           float* __restrict__ mean, float* __restrict__ rstd,
                           int C, int nimg, int hw) {
  __shared__ float s_sum[256];
  __shared__ float s_sq[256];
  int c = blockIdx.x;
  long per = (long)nimg * hw;
  float sum = 0.f, sq = 0.f;
  for (long i = threadIdx.x; i < per; i += blockDim.x) {
    int n = (int)(i / hw);
    int sp = (int)(i - (long)n * hw);
    float v = act[((long)n * C + c) * hw + sp];
    sum += v; sq += v * v;
  }
  s_sum[threadIdx.x] = sum; s_sq[threadIdx.x] = sq;
  __syncthreads();
  for (int s = 128; s > 0; s >>= 1) {
    if ((int)threadIdx.x < s) {
      s_sum[threadIdx.x] += s_sum[threadIdx.x + s];
      s_sq[threadIdx.x]  += s_sq[threadIdx.x + s];
    }
    __syncthreads();
  }
  if (threadIdx.x == 0) {
    float m = s_sum[0] / (float)per;
    float v = s_sq[0] / (float)per - m * m;
    mean[c] = m;
    rstd[c] = rsqrtf(v + 1e-5f);
  }
}

__global__ void k_bn_silu_bf16(const float* __restrict__ act,
                               const float* __restrict__ mean,
                               const float* __restrict__ rstd,
                               const float* __restrict__ g,
                               const float* __restrict__ b,
                               unsigned short* __restrict__ out,
                               long n, int C, int hw) {
  long i = (long)blockIdx.x * blockDim.x + threadIdx.x;
  if (i >= n) return;
  int c = (int)((i / hw) % C);
  float y = (act[i] - mean[c]) * rstd[c] * g[c] + b[c];
  float s = y / (1.0f + expf(-y));
  out[i] = f32_to_bf16_rn(s);
}

// ---------------------------------------------------------------------------
// Reparameterize / KL / tanh / recon-loss reductions
// ---------------------------------------------------------------------------
__global__ void k_reparam(const float* __restrict__ mean,
                          const float* __restrict__ logvar,
                          const float* __restrict__ noise,
                          float* __restrict__ z_f32,
                          unsigned short* __restrict__ z_bf, int n) {
  int i = blockIdx.x * blockDim.x + threadIdx.x;
  if (i >= n) return;
  float lv = logvar[i];
  float sp = (lv > 20.0f) ? lv : log1pf(expf(lv));
  float scale = sp + 1e-8f;
  float z = mean[i] + scale * noise[i];
  z_f32[i] = z;
  z_bf[i] = f32_to_bf16_rn(z);
}

__global__ void k_kl(const float* __restrict__ mean,
                     const float* __restrict__ logvar,
                     float* __restrict__ out, int n, float inv_batch) {
  __shared__ float sm[256];
  float acc = 0.f;
  for (int i = threadIdx.x; i < n; i += 256) {
    float lv = logvar[i];
    float sp = (lv > 20.0f) ? lv : log1pf(expf(lv));
    float s = sp + 1e-8f;
    float m = mean[i];
    acc += 0.5f * (s * s + m * m - 1.0f) - logf(s);
  }
  sm[threadIdx.x] = acc;
  __syncthreads();
  for (int s = 128; s > 0; s >>= 1) {
    if ((int)threadIdx.x < s) sm[threadIdx.x] += sm[threadIdx.x + s];
    __syncthreads();
  }
  if (threadIdx.x == 0) out[0] = sm[0] * inv_batch;
}

__global__ void k_tanh(const float* __restrict__ src, float* __restrict__ dst, long n) {
  long i = (long)blockIdx.x * blockDim.x + threadIdx.x;
  if (i < n) dst[i] = tanhf(src[i]);
}

__global__ void k_sqdiff_partial(const float* __restrict__ a,
                                 const float* __restrict__ b,
                                 float* __restrict__ partial, long n) {
  __shared__ float sm[256];
  float acc = 0.f;
  long stride = (long)gridDim.x * blockDim.x;
  for (long i = (long)blockIdx.x * blockDim.x + threadIdx.x; i < n; i += stride) {
    float d = a[i] - b[i];
    acc += d * d;
  }
  sm[threadIdx.x] = acc;
  __syncthreads();
  for (int s = 128; s > 0; s >>= 1) {
    if ((int)threadIdx.x < s) sm[threadIdx.x] += sm[threadIdx.x + s];
    __syncthreads();
  }
  if (threadIdx.x == 0) partial[blockIdx.x] = sm[0];
}

__global__ void k_final_losses(const float* __restrict__ partial, int nparts,
                               const float* __restrict__ kl,
                               float* __restrict__ out3, float inv_n) {
  __shared__ float sm[256];
  float acc = 0.f;
  for (int i = threadIdx.x; i < nparts; i += 256) acc += partial[i];
  sm[threadIdx.x] = acc;
  __syncthreads();
  for (int s = 128; s > 0; s >>= 1) {
    if ((int)threadIdx.x < s) sm[threadIdx.x] += sm[threadIdx.x + s];
    __syncthreads();
  }
  if (threadIdx.x == 0) {
    float lr = sm[0] * inv_n;
    float lk = kl[0];
    out3[0] = lr + lk;
    out3[1] = lr;
    out3[2] = lk;
  }
}

// ---------------------------------------------------------------------------
// Host orchestration
// ---------------------------------------------------------------------------
static inline unsigned eltBlocks(long n) { return (unsigned)((n + 255) / 256); }

extern "C" void kernel_launch(void* const* d_in, const int* in_sizes, int n_in,
                              void* d_out, int out_size, void* d_ws, size_t ws_size,
                              hipStream_t stream) {
  (void)in_sizes; (void)n_in; (void)out_size; (void)ws_size;

  const float* x     = (const float*)d_in[0];
  const float* noise = (const float*)d_in[1];
  const float *ew[6], *eb[6], *eg[6], *ebn[6];
  for (int i = 0; i < 6; ++i) {
    ew[i]  = (const float*)d_in[2 + 4 * i];
    eb[i]  = (const float*)d_in[3 + 4 * i];
    eg[i]  = (const float*)d_in[4 + 4 * i];
    ebn[i] = (const float*)d_in[5 + 4 * i];
  }
  const float* mw  = (const float*)d_in[26];
  const float* mb  = (const float*)d_in[27];
  const float* vw  = (const float*)d_in[28];
  const float* vb  = (const float*)d_in[29];
  const float* dwt = (const float*)d_in[30];
  const float* dbv = (const float*)d_in[31];
  const float *dtw[6], *dtb[6], *dg[5], *dbn[5];
  for (int i = 0; i < 5; ++i) {
    dtw[i] = (const float*)d_in[32 + 4 * i];
    dtb[i] = (const float*)d_in[33 + 4 * i];
    dg[i]  = (const float*)d_in[34 + 4 * i];
    dbn[i] = (const float*)d_in[35 + 4 * i];
  }
  dtw[5] = (const float*)d_in[52];
  dtb[5] = (const float*)d_in[53];

  char* wsb = (char*)d_ws;
  size_t off = 0;
  auto walloc = [&](size_t bytes) -> void* {
    void* p = wsb + off;
    off = (off + bytes + 255) & ~(size_t)255;
    return p;
  };
  const long NX = 16L * 3 * 512 * 512;
  const long NZ = 4194304;   // zero arena: 4M bf16 elems (>= max ci*ih*iw drift)

  // bf16 arena: zero arena first, then every buffer used as conv-gather input
  unsigned short* zbuf   = (unsigned short*)walloc(NZ * 2);
  unsigned short* x_bf   = (unsigned short*)walloc(NX * 2);
  unsigned short* bf_a   = (unsigned short*)walloc(33554432L * 2);
  unsigned short* bf_b   = (unsigned short*)walloc(33554432L * 2);
  unsigned short* d_bf   = (unsigned short*)walloc(65536L * 2);

  float*          act_f  = (float*)walloc(33554432L * 4);
  unsigned short* wconv  = (unsigned short*)walloc(65536L * 2);   // 64 x 1024 max
  unsigned short* wfc    = (unsigned short*)walloc(16777216L * 2);
  float*          mean_f = (float*)walloc(65536L * 4);
  float*          lvar_f = (float*)walloc(65536L * 4);
  float*          dvec_f = (float*)walloc(65536L * 4);
  unsigned short* z_bf   = (unsigned short*)walloc(65536L * 2);
  float*          bn_m   = (float*)walloc(64 * 4);
  float*          bn_r   = (float*)walloc(64 * 4);
  float*          kl_s   = (float*)walloc(4);
  float*          parts  = (float*)walloc(1024 * 4);

  k_zero_u64<<<eltBlocks(NZ / 4), 256, 0, stream>>>((unsigned long long*)zbuf, NZ / 4);
  k_pack_bf16<<<eltBlocks(NX), 256, 0, stream>>>(x, x_bf, NX);

  // ---- encoder: conv k3s2p1 -> BN -> SiLU (implicit-GEMM WMMA)
  const int e_ci[6] = {3, 32, 64, 64, 64, 64};
  const int e_co[6] = {32, 64, 64, 64, 64, 64};
  const int e_ih[6] = {512, 256, 128, 64, 32, 16};
  const unsigned short* cur = x_bf;
  unsigned short* nxt = bf_a;
  unsigned short* oth = bf_b;
  for (int i = 0; i < 6; ++i) {
    int ci = e_ci[i], co = e_co[i], ih = e_ih[i], iw = ih;
    int oh = ih / 2, ow = oh;
    int cip  = ci + (ci & 1);         // even ci so Kpad % 32 == 0
    int Kpad = cip * 16;              // k = ci*16 + ky*4 + kx
    int Mpad = (co + 15) & ~15;
    int N = 16 * oh * ow;
    int zoff = (int)(zbuf - cur);     // negative elem offset into zero arena
    k_pack_w_pad<<<eltBlocks((long)Mpad * Kpad), 256, 0, stream>>>(ew[i], wconv, co, ci, Mpad, Kpad, 0);
    dim3 g((unsigned)(N / 256), (unsigned)(Mpad / 16));
    k_wmma_gemm<1><<<g, 256, 0, stream>>>(wconv, cur, eb[i], 0, act_f,
                                          co, N, Kpad, ci, ih, iw, oh, ow, zoff);
    k_bn_stats<<<co, 256, 0, stream>>>(act_f, bn_m, bn_r, co, 16, oh * ow);
    long tot = 16L * co * oh * ow;
    k_bn_silu_bf16<<<eltBlocks(tot), 256, 0, stream>>>(act_f, bn_m, bn_r, eg[i], ebn[i],
                                                       nxt, tot, co, oh * ow);
    cur = nxt;
    unsigned short* t = nxt; nxt = oth; oth = t;
  }

  // ---- FC heads (M=16, K=N=4096); weights transposed to column-major bf16
  dim3 gfc(4096 / 256, 1);
  dim3 tpg(4096 / 32, 4096 / 32), tpb(32, 8);
  k_transpose_pack<<<tpg, tpb, 0, stream>>>(mw, wfc, 4096, 4096);
  k_wmma_gemm<0><<<gfc, 256, 0, stream>>>(cur, wfc, mb, 1, mean_f,
                                          16, 4096, 4096, 0, 0, 0, 0, 0, 0);
  k_transpose_pack<<<tpg, tpb, 0, stream>>>(vw, wfc, 4096, 4096);
  k_wmma_gemm<0><<<gfc, 256, 0, stream>>>(cur, wfc, vb, 1, lvar_f,
                                          16, 4096, 4096, 0, 0, 0, 0, 0, 0);

  float* z_out = (float*)d_out;
  k_reparam<<<eltBlocks(65536), 256, 0, stream>>>(mean_f, lvar_f, noise, z_out, z_bf, 65536);
  k_kl<<<1, 256, 0, stream>>>(mean_f, lvar_f, kl_s, 65536, 1.0f / 16.0f);

  k_transpose_pack<<<tpg, tpb, 0, stream>>>(dwt, wfc, 4096, 4096);
  k_wmma_gemm<0><<<gfc, 256, 0, stream>>>(z_bf, wfc, dbv, 1, dvec_f,
                                          16, 4096, 4096, 0, 0, 0, 0, 0, 0);
  k_pack_bf16<<<eltBlocks(65536), 256, 0, stream>>>(dvec_f, d_bf, 65536L);

  // ---- decoder: convT -> BN -> SiLU (x5) + convT -> tanh
  const int d_ci[6] = {64, 64, 64, 64, 32, 32};
  const int d_co[6] = {64, 64, 64, 32, 32, 3};
  const int d_ih[6] = {8, 16, 32, 64, 128, 256};
  float* xr = (float*)d_out + 65536;
  cur = d_bf; nxt = bf_a; oth = bf_b;
  for (int i = 0; i < 6; ++i) {
    int ci = d_ci[i], co = d_co[i], ih = d_ih[i], iw = ih;
    int oh = ih * 2, ow = oh;
    int cip  = ci + (ci & 1);
    int Kpad = cip * 16;
    int Mpad = (co + 15) & ~15;
    int N = 16 * oh * ow;
    int zoff = (int)(zbuf - cur);
    k_pack_w_pad<<<eltBlocks((long)Mpad * Kpad), 256, 0, stream>>>(dtw[i], wconv, co, ci, Mpad, Kpad, 1);
    dim3 g((unsigned)(N / 256), (unsigned)(Mpad / 16));
    k_wmma_gemm<2><<<g, 256, 0, stream>>>(wconv, cur, dtb[i], 0, act_f,
                                          co, N, Kpad, ci, ih, iw, oh, ow, zoff);
    if (i < 5) {
      k_bn_stats<<<co, 256, 0, stream>>>(act_f, bn_m, bn_r, co, 16, oh * ow);
      long tot = 16L * co * oh * ow;
      k_bn_silu_bf16<<<eltBlocks(tot), 256, 0, stream>>>(act_f, bn_m, bn_r, dg[i], dbn[i],
                                                         nxt, tot, co, oh * ow);
      cur = nxt;
      unsigned short* t = nxt; nxt = oth; oth = t;
    } else {
      k_tanh<<<eltBlocks(NX), 256, 0, stream>>>(act_f, xr, NX);
    }
  }

  // ---- losses
  k_sqdiff_partial<<<1024, 256, 0, stream>>>(xr, x, parts, NX);
  float* out3 = (float*)d_out + 65536 + NX;
  k_final_losses<<<1, 256, 0, stream>>>(parts, 1024, kl_s, out3, 1.0f / (float)NX);
}
